// LSTM_11390253269496
// MI455X (gfx1250) — matrix operand order repaired and checked by reading, not compiled, not run on hardware
//
#include <hip/hip_runtime.h>
#include <stdint.h>
#include <stddef.h>

// ---------------- problem constants ----------------
#define T_STEPS 2048
#define BATCH   64
#define I_DIM   77
#define KPAD    96          // 77 padded to 3 x 32 for bf16 WMMA
#define H_DIM   128
#define G_DIM   512         // 4*H
#define MROWS   (T_STEPS * BATCH)   // 131072

typedef __attribute__((ext_vector_type(16))) __bf16 v16bf;
typedef __attribute__((ext_vector_type(8)))  float  v8f;

union ABfrag { v16bf v; unsigned int u[8]; };

__device__ __forceinline__ unsigned short f2bf_bits(float f) {
  __bf16 h = (__bf16)f;
  return __builtin_bit_cast(unsigned short, h);
}
__device__ __forceinline__ float sigmoid_f(float x) {
  return 1.0f / (1.0f + __expf(-x));
}

// ---------------- prep kernels ----------------

// inputs [T*B, 77] f32 -> [T*B, 96] bf16 (zero padded K)
__global__ void prep_inputs(const float* __restrict__ x,
                            __bf16* __restrict__ out) {
  int idx = blockIdx.x * blockDim.x + threadIdx.x;
  if (idx >= MROWS * KPAD) return;
  int row = idx / KPAD;
  int k   = idx - row * KPAD;
  float v = (k < I_DIM) ? x[(size_t)row * I_DIM + k] : 0.0f;
  out[idx] = (__bf16)v;
}

// W_ih [512,77] -> swizzled bf16 B-operand [(k/2)*1024 + n*2 + (k&1)], k<96 padded
__global__ void prep_wih(const float* __restrict__ W,
                         __bf16* __restrict__ out) {
  int idx = blockIdx.x * blockDim.x + threadIdx.x;   // KPAD*G_DIM
  if (idx >= KPAD * G_DIM) return;
  int k = idx / G_DIM;
  int n = idx - k * G_DIM;
  float v = (k < I_DIM) ? W[(size_t)n * I_DIM + k] : 0.0f;
  out[(k >> 1) * (2 * G_DIM) + n * 2 + (k & 1)] = (__bf16)v;
}

// W_hh [512,128] -> swizzled bf16 B-operand, K = 128
__global__ void prep_whh(const float* __restrict__ W,
                         __bf16* __restrict__ out) {
  int idx = blockIdx.x * blockDim.x + threadIdx.x;   // H_DIM*G_DIM
  if (idx >= H_DIM * G_DIM) return;
  int k = idx / G_DIM;
  int n = idx - k * G_DIM;
  float v = W[(size_t)n * H_DIM + k];
  out[(k >> 1) * (2 * G_DIM) + n * 2 + (k & 1)] = (__bf16)v;
}

__global__ void prep_bias(const float* __restrict__ b_ih,
                          const float* __restrict__ b_hh,
                          float* __restrict__ bias) {
  int idx = blockIdx.x * blockDim.x + threadIdx.x;
  if (idx < G_DIM) bias[idx] = b_ih[idx] + b_hh[idx];
}

// ---------------- x_proj GEMM:  xp = inA(bf16) * Wih + bias ----------------
// 256 threads = 8 waves; each wave owns one 16-row M tile, sweeps all 32 N tiles.
__global__ __launch_bounds__(256) void xproj_kernel(
    const __bf16* __restrict__ A,      // [MROWS, KPAD] bf16
    const __bf16* __restrict__ Bsw,    // swizzled [KPAD, G_DIM]
    const float*  __restrict__ bias,   // [G_DIM]
    float* __restrict__ xp) {          // [MROWS, G_DIM] f32
  __shared__ unsigned int Bl[(KPAD / 2) * G_DIM];   // 24576 dwords = 96 KB
  __shared__ float bias_l[G_DIM];

  const int tid = threadIdx.x;
  const unsigned int* Bg = (const unsigned int*)Bsw;
#pragma unroll
  for (int i = 0; i < (KPAD / 2) * G_DIM / 256; ++i)
    Bl[i * 256 + tid] = Bg[i * 256 + tid];
  if (tid < G_DIM / 2) { bias_l[tid] = bias[tid]; bias_l[tid + 256] = bias[tid + 256]; }
  __syncthreads();

  const int wave  = tid >> 5;
  const int lane  = tid & 31;
  const int mtile = blockIdx.x * 8 + wave;
  const int m0    = mtile * 16;
  const int row   = m0 + (lane & 15);
  const int khalf = lane >> 4;
  const int ncol  = lane & 15;

  const unsigned int* Aru = (const unsigned int*)A;
  ABfrag a[3];
#pragma unroll
  for (int kc = 0; kc < 3; ++kc)
#pragma unroll
    for (int v = 0; v < 8; ++v) {
      int khalfidx = kc * 16 + 8 * (v >> 2) + 4 * khalf + (v & 3);   // K/2
      a[kc].u[v] = Aru[(size_t)row * (KPAD / 2) + khalfidx];
    }

  for (int nt = 0; nt < 32; ++nt) {
    const int col = nt * 16 + ncol;
    float bb = bias_l[col];
    v8f acc;
#pragma unroll
    for (int r = 0; r < 8; ++r) acc[r] = bb;
#pragma unroll
    for (int kc = 0; kc < 3; ++kc) {
      ABfrag b;
#pragma unroll
      for (int v = 0; v < 8; ++v)
        b.u[v] = Bl[(kc * 16 + 8 * (v >> 2) + 4 * khalf + (v & 3)) * G_DIM + col];
      acc = __builtin_amdgcn_wmma_f32_16x16x32_bf16(
          false, a[kc].v, false, b.v, (short)0, acc, false, false);
    }
#pragma unroll
    for (int r = 0; r < 8; ++r)
      xp[(size_t)(m0 + r + 8 * khalf) * G_DIM + col] = acc[r];
  }
}

// ---------------- persistent single-WGP LSTM recurrence ----------------
// 1024 threads = 32 waves on one WGP. All state lives in LDS (~305 KB of 320 KB).
__global__ __launch_bounds__(1024) void lstm_rec(
    const __bf16* __restrict__ Whh_sw,   // swizzled [H_DIM, G_DIM] bf16
    const float*  __restrict__ xp,       // [T, B, G_DIM] f32
    const float*  __restrict__ Wout,     // [H_DIM]
    const float*  __restrict__ b_out,    // [1]
    float* __restrict__ out) {           // [T, B]
  __shared__ unsigned int Wlds[(H_DIM / 2) * G_DIM];   // 128 KB
  __shared__ float gates[BATCH * G_DIM];               // 128 KB (reused for logit partials)
  __shared__ unsigned int hbfU[BATCH * H_DIM / 2];     // 16 KB: h as bf16 pairs
  __shared__ float c_lds[BATCH * H_DIM];               // 32 KB
  __shared__ float wout_l[H_DIM];

  const int tid = threadIdx.x;

  // stage W_hh into LDS (coalesced dwords)
  const unsigned int* Wg = (const unsigned int*)Whh_sw;
#pragma unroll
  for (int i = 0; i < (H_DIM / 2) * G_DIM / 1024; ++i)
    Wlds[i * 1024 + tid] = Wg[i * 1024 + tid];
  if (tid < H_DIM) wout_l[tid] = Wout[tid];
  // h0 = c0 = 1.0
  unsigned short* hbf = (unsigned short*)hbfU;
#pragma unroll
  for (int p = 0; p < 8; ++p) {
    int q = p * 1024 + tid;
    hbf[q] = 0x3F80;        // bf16(1.0)
    c_lds[q] = 1.0f;
  }
  const float bo = b_out[0];
  __syncthreads();

  const int wave  = tid >> 5;
  const int lane  = tid & 31;
  const int mtile = wave & 3;            // 4 M tiles (B = 64)
  const int ngrp  = wave >> 2;           // 8 groups of 4 N tiles
  const int m0    = mtile * 16;
  const int row   = m0 + (lane & 15);
  const int khalf = lane >> 4;
  const int ncol  = lane & 15;
  const int hcol  = tid & 127;           // owned h' in update phase

  for (int t = 0; t < T_STEPS; ++t) {
    // ---- A fragments from h (bf16, LDS), 4 K-chunks ----
    ABfrag a[4];
#pragma unroll
    for (int kc = 0; kc < 4; ++kc)
#pragma unroll
      for (int v = 0; v < 8; ++v) {
        int khalfidx = kc * 16 + 8 * (v >> 2) + 4 * khalf + (v & 3);  // K/2
        a[kc].u[v] = hbfU[row * (H_DIM / 2) + khalfidx];
      }

    // ---- gate GEMM: gates = xp[t] + h @ Whh^T, fused activations ----
    const float* xpt = xp + (size_t)t * (BATCH * G_DIM);
#pragma unroll
    for (int i = 0; i < 4; ++i) {
      const int nt  = ngrp * 4 + i;
      const int col = nt * 16 + ncol;
      v8f acc;
#pragma unroll
      for (int r = 0; r < 8; ++r)
        acc[r] = xpt[(m0 + r + 8 * khalf) * G_DIM + col];
#pragma unroll
      for (int kc = 0; kc < 4; ++kc) {
        ABfrag b;
#pragma unroll
        for (int v = 0; v < 8; ++v)
          b.u[v] = Wlds[(kc * 16 + 8 * (v >> 2) + 4 * khalf + (v & 3)) * G_DIM + col];
        acc = __builtin_amdgcn_wmma_f32_16x16x32_bf16(
            false, a[kc].v, false, b.v, (short)0, acc, false, false);
      }
      const int gate = nt >> 3;          // 0:i 1:f 2:g 3:o
#pragma unroll
      for (int r = 0; r < 8; ++r) {
        float x = acc[r];
        float y = (gate == 2) ? tanhf(x) : sigmoid_f(x);
        gates[(m0 + r + 8 * khalf) * G_DIM + col] = y;
      }
    }
    // prefetch next timestep's x_proj slice into the cache hierarchy
    if (t + 1 < T_STEPS)
      __builtin_prefetch(xpt + (BATCH * G_DIM) + m0 * G_DIM + ngrp * 64, 0, 1);
    __syncthreads();

    // ---- elementwise cell update; each thread owns 8 (b,h') pairs ----
#pragma unroll
    for (int p = 0; p < 8; ++p) {
      int q  = p * 1024 + tid;           // q = b*128 + h'
      int b_ = q >> 7;
      float ig = gates[b_ * G_DIM + hcol];
      float fg = gates[b_ * G_DIM + 128 + hcol];
      float gg = gates[b_ * G_DIM + 256 + hcol];
      float og = gates[b_ * G_DIM + 384 + hcol];
      float c  = fg * c_lds[q] + ig * gg;
      c_lds[q] = c;
      float h  = og * tanhf(c);
      hbf[q]   = f2bf_bits(h);
      // deterministic logit partial: reuse own 'i' gate slot (read-complete above)
      gates[b_ * G_DIM + hcol] = h * wout_l[hcol];
    }
    __syncthreads();

    // ---- fixed-order logit reduction (deterministic) ----
    if (tid < BATCH) {
      float s = bo;
#pragma unroll 8
      for (int j = 0; j < H_DIM; ++j) s += gates[tid * G_DIM + j];
      out[t * BATCH + tid] = s;
    }
    __syncthreads();
  }
}

// ---------------- host launch ----------------
extern "C" void kernel_launch(void* const* d_in, const int* in_sizes, int n_in,
                              void* d_out, int out_size, void* d_ws, size_t ws_size,
                              hipStream_t stream) {
  const float* inputs = (const float*)d_in[0];   // [T,B,I]
  const float* W_ih   = (const float*)d_in[1];   // [4H,I]
  const float* W_hh   = (const float*)d_in[2];   // [4H,H]
  const float* b_ih   = (const float*)d_in[3];   // [4H]
  const float* b_hh   = (const float*)d_in[4];   // [4H]
  const float* W_out  = (const float*)d_in[5];   // [O,H]
  const float* b_out  = (const float*)d_in[6];   // [O]
  float* out = (float*)d_out;

  char* ws = (char*)d_ws;
  const size_t off_xp   = 0;
  const size_t off_inA  = off_xp  + (size_t)MROWS * G_DIM * sizeof(float);   // 256 MB
  const size_t off_wih  = off_inA + (size_t)MROWS * KPAD * sizeof(short);    // +24 MB
  const size_t off_whh  = off_wih + (size_t)KPAD * G_DIM * sizeof(short);
  const size_t off_bias = off_whh + (size_t)H_DIM * G_DIM * sizeof(short);

  float*  xp   = (float*)(ws + off_xp);
  __bf16* inA  = (__bf16*)(ws + off_inA);
  __bf16* wih  = (__bf16*)(ws + off_wih);
  __bf16* whh  = (__bf16*)(ws + off_whh);
  float*  bias = (float*)(ws + off_bias);

  prep_inputs<<<(MROWS * KPAD + 255) / 256, 256, 0, stream>>>(inputs, inA);
  prep_wih<<<(KPAD * G_DIM + 255) / 256, 256, 0, stream>>>(W_ih, wih);
  prep_whh<<<(H_DIM * G_DIM + 255) / 256, 256, 0, stream>>>(W_hh, whh);
  prep_bias<<<2, 256, 0, stream>>>(b_ih, b_hh, bias);

  xproj_kernel<<<MROWS / 16 / 8, 256, 0, stream>>>(inA, wih, bias, xp);

  lstm_rec<<<1, 1024, 0, stream>>>(whh, xp, W_out, b_out, out);
}
